// MultiView_GAT_26096221290528
// MI455X (gfx1250) — matrix operand order repaired and checked by reading, not compiled, not run on hardware
//
#include <hip/hip_runtime.h>

#define CIN   128   // IN == HID
#define HIDC  128
#define OUTC  64
#define NVIEW 3
#define NEG_SLOPE 0.2f
#define WCAP  (HIDC * CIN)   // bf16 elements reserved per weight matrix

typedef __bf16 v16bf __attribute__((ext_vector_type(16)));
typedef __bf16 v8bf  __attribute__((ext_vector_type(8)));
typedef float  v8f   __attribute__((ext_vector_type(8)));

static __device__ __forceinline__ __bf16 f2bf(float f) {
    unsigned u = __float_as_uint(f);
    u += 0x7FFFu + ((u >> 16) & 1u);           // round-to-nearest-even
    unsigned short h = (unsigned short)(u >> 16);
    return __builtin_bit_cast(__bf16, h);
}

// order-preserving float <-> u32 key (for atomicMax-based segment max)
static __device__ __forceinline__ unsigned fkey(float f) {
    unsigned u = __float_as_uint(f);
    return (u & 0x80000000u) ? ~u : (u | 0x80000000u);
}
static __device__ __forceinline__ float keyf(unsigned k) {
    unsigned u = (k & 0x80000000u) ? (k & 0x7FFFFFFFu) : ~k;
    return __uint_as_float(u);
}

// two contiguous 16-byte bf16 chunks -> one WMMA operand
static __device__ __forceinline__ v16bf ld16(const __bf16* p0, const __bf16* p1) {
    v8bf a = *(const v8bf*)p0;
    v8bf b = *(const v8bf*)p1;
    v16bf r;
#pragma unroll
    for (int i = 0; i < 8; ++i) { r[i] = a[i]; r[i + 8] = b[i]; }
    return r;
}

// ---------------------------------------------------------------------------
// f32 -> bf16 staging kernels
// ---------------------------------------------------------------------------
__global__ void conv_bf16(const float* __restrict__ in, __bf16* __restrict__ out,
                          long long n) {
    long long i = (long long)blockIdx.x * blockDim.x + threadIdx.x;
    if (i < n) out[i] = f2bf(in[i]);
}

__global__ void conv_weights(const float* __restrict__ Wl,
                             const float* __restrict__ Wr,
                             const float* __restrict__ Wres,
                             __bf16* __restrict__ wbf, int n_each) {
    int i = blockIdx.x * blockDim.x + threadIdx.x;
    if (i >= n_each) return;
    wbf[i]        = f2bf(Wl[i]);
    wbf[WCAP + i] = f2bf(Wr[i]);
    if (Wres) wbf[2 * WCAP + i] = f2bf(Wres[i]);
}

// ---------------------------------------------------------------------------
// Fused multi-output GEMM on pre-staged bf16: ol=h@Wl^T, or=h@Wr^T,
// (ores=h@Wres^T if HAS_RES). One wave -> one 16x16 f32 tile; K-loop of 4
// v_wmma_f32_16x16x32_bf16 per B matrix; all operand chunks are contiguous
// 16-byte b128 loads.  HAS_RES is a template arg -> branch-free inner loop.
// ---------------------------------------------------------------------------
template <int HAS_RES>
__global__ void gat_gemm_wmma(const __bf16* __restrict__ hbf,
                              const __bf16* __restrict__ wbf,   // [Wl|Wr|Wres] @ WCAP stride
                              float* __restrict__ ol,
                              float* __restrict__ orr,
                              float* __restrict__ ores,
                              int nrows, int cout) {
    const int lane  = threadIdx.x & 31;
    const int wave  = threadIdx.x >> 5;
    const int mtile = blockIdx.x * (blockDim.x >> 5) + wave;
    const int m0    = mtile * 16;
    if (m0 >= nrows) return;
    const int n0  = blockIdx.y * 16;
    const int mlo = lane & 15;          // A row / B col / D col
    const int kh  = lane >> 4;          // K-half selector

    int arow = m0 + mlo;
    if (arow >= nrows) arow = nrows - 1;          // only affects never-stored rows
    const __bf16* ap = hbf + (size_t)arow * CIN + (kh << 4);
    const int col    = n0 + mlo;
    const __bf16* bl = wbf + (size_t)col * CIN + (kh << 4);
    const __bf16* br = bl + WCAP;
    const __bf16* bs = bl + 2 * WCAP;

    v8f cl = {}; v8f cr = {}; v8f cs = {};

#pragma unroll
    for (int k0 = 0; k0 < CIN; k0 += 32) {
        const v16bf a  = ld16(ap + k0, ap + k0 + 8);
        const v16bf vl = ld16(bl + k0, bl + k0 + 8);
        const v16bf vr = ld16(br + k0, br + k0 + 8);
        cl = __builtin_amdgcn_wmma_f32_16x16x32_bf16(false, a, false, vl,
                                                     (short)0, cl, false, false);
        cr = __builtin_amdgcn_wmma_f32_16x16x32_bf16(false, a, false, vr,
                                                     (short)0, cr, false, false);
        if (HAS_RES) {
            const v16bf vs = ld16(bs + k0, bs + k0 + 8);
            cs = __builtin_amdgcn_wmma_f32_16x16x32_bf16(false, a, false, vs,
                                                         (short)0, cs, false, false);
        }
    }
#pragma unroll
    for (int r = 0; r < 8; ++r) {
        const int row = m0 + (kh << 3) + r;
        if (row < nrows) {
            ol [(size_t)row * cout + col] = cl[r];
            orr[(size_t)row * cout + col] = cr[r];
            if (HAS_RES) ores[(size_t)row * cout + col] = cs[r];
        }
    }
}

// ---------------------------------------------------------------------------
// Edge pass 1: alpha[e] = att . leaky_relu(hr[dst] + hl[src] + ea*We),
// segment-max over dst via ordered-u32 atomicMax. One wave32 per edge.
// ---------------------------------------------------------------------------
__global__ void gat_edge_logits(const float* __restrict__ hl,
                                const float* __restrict__ hr,
                                const float* __restrict__ Wev,
                                const float* __restrict__ att,
                                const int* __restrict__ src,
                                const int* __restrict__ dst,
                                const float* __restrict__ ea,
                                float* __restrict__ alpha,
                                unsigned* __restrict__ amaxk,
                                int nedges, int C) {
    const int wid  = (int)((blockIdx.x * blockDim.x + threadIdx.x) >> 5);
    const int lane = threadIdx.x & 31;
    if (wid >= nedges) return;
    const int s = src[wid];
    const int d = dst[wid];
    const float eav = ea[wid];
    float p = 0.0f;
    for (int c = lane; c < C; c += 32) {
        float t = hr[(size_t)d * C + c] + hl[(size_t)s * C + c] + eav * Wev[c];
        t = (t > 0.0f) ? t : NEG_SLOPE * t;
        p += t * att[c];
    }
#pragma unroll
    for (int off = 16; off > 0; off >>= 1) p += __shfl_down(p, off, 32);
    if (lane == 0) {
        alpha[wid] = p;
        atomicMax(&amaxk[d], fkey(p));
    }
}

// ---------------------------------------------------------------------------
// Edge pass 2: ae = exp(alpha - amax[dst]); accum[dst] += ae * hl[src];
// denom[dst] += ae.  Normalization deferred to the node pass.
// ---------------------------------------------------------------------------
__global__ void gat_edge_scatter(const float* __restrict__ hl,
                                 const float* __restrict__ alpha,
                                 const unsigned* __restrict__ amaxk,
                                 const int* __restrict__ src,
                                 const int* __restrict__ dst,
                                 float* __restrict__ accum,
                                 float* __restrict__ denom,
                                 int nedges, int C) {
    const int wid  = (int)((blockIdx.x * blockDim.x + threadIdx.x) >> 5);
    const int lane = threadIdx.x & 31;
    if (wid >= nedges) return;
    const int s = src[wid];
    const int d = dst[wid];
    const float ae = __expf(alpha[wid] - keyf(amaxk[d]));
    for (int c = lane; c < C; c += 32)
        atomicAdd(&accum[(size_t)d * C + c], ae * hl[(size_t)s * C + c]);
    if (lane == 0) atomicAdd(&denom[d], ae);
}

// Hidden-layer finalize: divide, residual, bias, ELU -> bf16 (next GEMM input).
__global__ void gat_finalize_hidden(const float* __restrict__ accum,
                                    const float* __restrict__ denom,
                                    const float* __restrict__ hres,
                                    const float* __restrict__ bias,
                                    __bf16* __restrict__ hout,
                                    long long total, int C) {
    long long i = (long long)blockIdx.x * blockDim.x + threadIdx.x;
    if (i >= total) return;
    const int n = (int)(i / C);
    const int c = (int)(i % C);
    float v = accum[i] / (denom[n] + 1e-16f) + hres[i] + bias[c];
    v = (v > 0.0f) ? v : (__expf(v) - 1.0f);   // ELU
    hout[i] = f2bf(v);
}

// Output-layer finalize: divide + bias -> f32 view embedding.
__global__ void gat_finalize_out(const float* __restrict__ accum,
                                 const float* __restrict__ denom,
                                 const float* __restrict__ bias,
                                 float* __restrict__ emb,
                                 long long total, int C) {
    long long i = (long long)blockIdx.x * blockDim.x + threadIdx.x;
    if (i >= total) return;
    const int n = (int)(i / C);
    const int c = (int)(i % C);
    emb[i] = accum[i] / (denom[n] + 1e-16f) + bias[c];
}

__global__ void gat_zero(float* __restrict__ p, long long n) {
    long long i = (long long)blockIdx.x * blockDim.x + threadIdx.x;
    if (i < n) p[i] = 0.0f;
}

// ---------------------------------------------------------------------------
// Semantic attention across views; writes (global_embedding[N,64], weights[N,3]).
// ---------------------------------------------------------------------------
__global__ void gat_semantic(const float* __restrict__ embs,   // [V][N][OUTC]
                             const float* __restrict__ sem,    // [V][V*OUTC]
                             float* __restrict__ outg,         // [N][OUTC]
                             float* __restrict__ outw,         // [N][V]
                             int n) {
    const int i = blockIdx.x * blockDim.x + threadIdx.x;
    if (i >= n) return;
    float lg[NVIEW] = {0.0f, 0.0f, 0.0f};
    for (int v = 0; v < NVIEW; ++v) {
        const float* e = embs + ((size_t)v * n + i) * OUTC;
        for (int c = 0; c < OUTC; ++c) {
            const float f = e[c];
#pragma unroll
            for (int w = 0; w < NVIEW; ++w)
                lg[w] += f * sem[(size_t)w * (NVIEW * OUTC) + v * OUTC + c];
        }
    }
    const float m = fmaxf(lg[0], fmaxf(lg[1], lg[2]));
    float wt[NVIEW], s = 0.0f;
#pragma unroll
    for (int w = 0; w < NVIEW; ++w) { wt[w] = __expf(lg[w] - m); s += wt[w]; }
#pragma unroll
    for (int w = 0; w < NVIEW; ++w) wt[w] /= s;
    for (int c = 0; c < OUTC; ++c) {
        float g = 0.0f;
#pragma unroll
        for (int v = 0; v < NVIEW; ++v)
            g += wt[v] * embs[((size_t)v * n + i) * OUTC + c];
        outg[(size_t)i * OUTC + c] = g;
    }
#pragma unroll
    for (int w = 0; w < NVIEW; ++w) outw[(size_t)i * NVIEW + w] = wt[w];
}

extern "C" void kernel_launch(void* const* d_in, const int* in_sizes, int n_in,
                              void* d_out, int out_size, void* d_ws, size_t ws_size,
                              hipStream_t stream) {
    const float* x      = (const float*)d_in[0];
    const int*   ei     = (const int*)  d_in[1];
    const float* eattr  = (const float*)d_in[2];
    const float* Wl12   = (const float*)d_in[3];
    const float* Wr12   = (const float*)d_in[4];
    const float* We12   = (const float*)d_in[5];
    const float* att12  = (const float*)d_in[6];
    const float* b12    = (const float*)d_in[7];
    const float* Wres12 = (const float*)d_in[8];
    const float* Wl3    = (const float*)d_in[9];
    const float* Wr3    = (const float*)d_in[10];
    const float* We3    = (const float*)d_in[11];
    const float* att3   = (const float*)d_in[12];
    const float* b3     = (const float*)d_in[13];
    const float* sem    = (const float*)d_in[14];

    const int N = in_sizes[0] / CIN;          // 50000
    const int E = in_sizes[1] / (NVIEW * 2);  // 800000

    // f32 workspace: hl | hr | hres | accum | denom | amaxk | alpha | embs | <bf16 region>
    float* ws    = (float*)d_ws;
    const size_t NH = (size_t)N * HIDC;
    float* hl    = ws;
    float* hr    = ws + NH;
    float* hres  = ws + 2 * NH;
    float* accum = ws + 3 * NH;
    float* denom = ws + 4 * NH;
    unsigned* amaxk = (unsigned*)(ws + 4 * NH + (size_t)N);
    float* alphaB = ws + 4 * NH + 2 * (size_t)N;
    float* embs   = alphaB + (size_t)E;                       // [V][N][OUTC]
    // bf16 region: xbf | hbf0 | hbf1 | wbf[3*WCAP]
    __bf16* xbf  = (__bf16*)(embs + (size_t)NVIEW * N * OUTC);
    __bf16* hbf0 = xbf + NH;
    __bf16* hbf1 = hbf0 + NH;
    __bf16* wbf  = hbf1 + NH;

    const int mt = (N + 15) / 16;
    const dim3 gemmBlk(128);                               // 4 waves / block
    const int  gemmGX = (mt + 3) / 4;
    const int  edgeBlocks = (E + 7) / 8;                   // 8 waves of 32 per block
    const long long zeroN = (long long)NH + 2LL * N;       // accum + denom + amaxk
    const int zeroBlocks = (int)((zeroN + 255) / 256);
    const long long totH = (long long)N * HIDC;
    const long long totO = (long long)N * OUTC;

    // stage x -> bf16 once (shared across all views)
    conv_bf16<<<(int)((totH + 255) / 256), 256, 0, stream>>>(x, xbf, totH);

    for (int v = 0; v < NVIEW; ++v) {
        const int* srcv = ei + (size_t)v * 2 * E;
        const int* dstv = srcv + E;
        const float* eav = eattr + (size_t)v * E;
        const __bf16* hin = xbf;
        __bf16* houts[2] = {hbf0, hbf1};

        for (int l = 0; l < 2; ++l) {
            const size_t wo = (size_t)(v * 2 + l);
            conv_weights<<<(HIDC * CIN + 255) / 256, 256, 0, stream>>>(
                Wl12 + wo * HIDC * CIN, Wr12 + wo * HIDC * CIN,
                Wres12 + wo * HIDC * CIN, wbf, HIDC * CIN);
            gat_gemm_wmma<1><<<dim3(gemmGX, HIDC / 16), gemmBlk, 0, stream>>>(
                hin, wbf, hl, hr, hres, N, HIDC);
            gat_zero<<<zeroBlocks, 256, 0, stream>>>(accum, zeroN);
            gat_edge_logits<<<edgeBlocks, 256, 0, stream>>>(
                hl, hr, We12 + wo * HIDC, att12 + wo * HIDC,
                srcv, dstv, eav, alphaB, amaxk, E, HIDC);
            gat_edge_scatter<<<edgeBlocks, 256, 0, stream>>>(
                hl, alphaB, amaxk, srcv, dstv, accum, denom, E, HIDC);
            gat_finalize_hidden<<<(int)((totH + 255) / 256), 256, 0, stream>>>(
                accum, denom, hres, b12 + wo * HIDC, houts[l], totH, HIDC);
            hin = houts[l];
        }

        // output layer: cout = 64, no residual, no ELU
        conv_weights<<<(OUTC * CIN + 255) / 256, 256, 0, stream>>>(
            Wl3 + (size_t)v * OUTC * CIN, Wr3 + (size_t)v * OUTC * CIN,
            nullptr, wbf, OUTC * CIN);
        gat_gemm_wmma<0><<<dim3(gemmGX, OUTC / 16), gemmBlk, 0, stream>>>(
            hin, wbf, hl, hr, nullptr, N, OUTC);
        gat_zero<<<zeroBlocks, 256, 0, stream>>>(accum, zeroN);
        gat_edge_logits<<<edgeBlocks, 256, 0, stream>>>(
            hl, hr, We3 + (size_t)v * OUTC, att3 + (size_t)v * OUTC,
            srcv, dstv, eav, alphaB, amaxk, E, OUTC);
        gat_edge_scatter<<<edgeBlocks, 256, 0, stream>>>(
            hl, alphaB, amaxk, srcv, dstv, accum, denom, E, OUTC);
        gat_finalize_out<<<(int)((totO + 255) / 256), 256, 0, stream>>>(
            accum, denom, b3 + (size_t)v * OUTC,
            embs + (size_t)v * N * OUTC, totO, OUTC);
    }

    float* outg = (float*)d_out;               // [N, OUTC]
    float* outw = outg + (size_t)N * OUTC;     // [N, NVIEW]
    gat_semantic<<<(N + 255) / 256, 256, 0, stream>>>(embs, sem, outg, outw, N);
}
